// ComplexGAT_7447473291644
// MI455X (gfx1250) — matrix operand order repaired and checked by reading, not compiled, not run on hardware
//
#include <hip/hip_runtime.h>
#include <hip/hip_bf16.h>
#include <math.h>

#define N_NODES  50000
#define E_EDGES  800000
#define E_TOT    (E_EDGES + N_NODES)
#define G_GRAPHS 64
#define IN_DIM   512
#define HID      128
#define HEADS    2
#define HC       (HID * HEADS)   // 256
#define OUT_DIM  64
#define NEG_SLOPE 0.2f
#define BN_EPS    1e-5f
#define WT        4              // 16-col output tiles per wave

typedef __attribute__((ext_vector_type(16))) __bf16 v16bf;
typedef __attribute__((ext_vector_type(8)))  __bf16 v8bf;
typedef __attribute__((ext_vector_type(8)))  float  v8f;

// ---------------------------------------------------------------- utilities
__global__ void fill_f32(float* __restrict__ p, float v, long n) {
    long i = (long)blockIdx.x * blockDim.x + threadIdx.x;
    if (i < n) p[i] = v;
}

__device__ inline unsigned short f2bf_rne(float f) {
    unsigned int u = __float_as_uint(f);
    return (unsigned short)((u + 0x7FFFu + ((u >> 16) & 1u)) >> 16);
}

__global__ void f32_to_bf16(const float* __restrict__ in, unsigned short* __restrict__ out, long n) {
    long i = (long)blockIdx.x * blockDim.x + threadIdx.x;
    if (i < n) out[i] = f2bf_rne(in[i]);
}

// Convert W (fp32, [K,Ncol] row-major) into bf16 WMMA B-fragment order:
// flat index (((nt*KB + kb)*32 + lane)*16 + e) holds B[kb*32 + (lane>>4)*16 + e, nt*16 + (lane&15)]
// so the GEMM's per-lane B fragment is one contiguous 32-byte load.
__global__ void pack_W_bf16(const float* __restrict__ W, unsigned short* __restrict__ Bp,
                            int K, int Ncol)
{
    long i = (long)blockIdx.x * blockDim.x + threadIdx.x;
    long total = (long)K * Ncol;
    if (i >= total) return;
    int e = (int)(i & 15);
    int l = (int)((i >> 4) & 31);
    long rest = i >> 9;
    int KB = K >> 5;
    int kb = (int)(rest % KB);
    int nt = (int)(rest / KB);
    int k = kb * 32 + (l >> 4) * 16 + e;
    int n = nt * 16 + (l & 15);
    Bp[i] = f2bf_rne(W[(size_t)k * Ncol + n]);
}

// ------------------------------------------------------------ WMMA bf16 GEMM
// C[M,Ncol] = A[M,K] * Bpacked, A bf16 row-major, C fp32.
// Each wave computes a 16x64 strip (4 accumulators sharing one A fragment).
// blockDim.x = 32 * (Ncol/64); one block covers the full output width.
__global__ __launch_bounds__(128) void gemm_bf16_wmma(
    const unsigned short* __restrict__ A, const unsigned short* __restrict__ Bp,
    float* __restrict__ C, int M, int K, int Ncol)
{
    const int wave = threadIdx.x >> 5;
    const int lane = threadIdx.x & 31;
    const int grp  = lane >> 4;        // lane half (0: lanes 0-15, 1: lanes 16-31)
    const int lm   = lane & 15;
    const int m0   = blockIdx.x * 16;
    const int nt0  = wave * WT;        // first of 4 consecutive 16-col tiles
    const int KB   = K >> 5;

    const __bf16* Ab = reinterpret_cast<const __bf16*>(A);
    const v16bf*  Bf = reinterpret_cast<const v16bf*>(Bp);

    v8f acc0 = {}, acc1 = {}, acc2 = {}, acc3 = {};
    for (int kb = 0; kb < KB; ++kb) {
        // A fragment: row m0+lm; lane holds K = kb*32 + grp*16 + [0..15]
        const __bf16* arow = Ab + (size_t)(m0 + lm) * K + kb * 32 + grp * 16;
        v8bf alo = *reinterpret_cast<const v8bf*>(arow);
        v8bf ahi = *reinterpret_cast<const v8bf*>(arow + 8);
        v16bf a;
#pragma unroll
        for (int e = 0; e < 8; ++e) { a[e] = alo[e]; a[8 + e] = ahi[e]; }

        v16bf b0 = Bf[((size_t)(nt0 + 0) * KB + kb) * 32 + lane];
        v16bf b1 = Bf[((size_t)(nt0 + 1) * KB + kb) * 32 + lane];
        v16bf b2 = Bf[((size_t)(nt0 + 2) * KB + kb) * 32 + lane];
        v16bf b3 = Bf[((size_t)(nt0 + 3) * KB + kb) * 32 + lane];

        acc0 = __builtin_amdgcn_wmma_f32_16x16x32_bf16(false, a, false, b0, (short)0, acc0, false, false);
        acc1 = __builtin_amdgcn_wmma_f32_16x16x32_bf16(false, a, false, b1, (short)0, acc1, false, false);
        acc2 = __builtin_amdgcn_wmma_f32_16x16x32_bf16(false, a, false, b2, (short)0, acc2, false, false);
        acc3 = __builtin_amdgcn_wmma_f32_16x16x32_bf16(false, a, false, b3, (short)0, acc3, false, false);
    }
    // D layout: VGPR j holds row m0 + grp*8 + j, col ntile*16 + lm
    float* crow = C + (size_t)(m0 + grp * 8) * Ncol + lm;
#pragma unroll
    for (int j = 0; j < 8; ++j) {
        crow[(size_t)j * Ncol + (nt0 + 0) * 16] = acc0[j];
        crow[(size_t)j * Ncol + (nt0 + 1) * 16] = acc1[j];
        crow[(size_t)j * Ncol + (nt0 + 2) * 16] = acc2[j];
        crow[(size_t)j * Ncol + (nt0 + 3) * 16] = acc3[j];
    }
}

// --------------------------------------------------------------- GAT pieces
__global__ void gat_scores(const float* __restrict__ xl, const float* __restrict__ asrc,
                           const float* __restrict__ adst, float* __restrict__ es,
                           float* __restrict__ ed, int H, int C)
{
    int i = blockIdx.x * blockDim.x + threadIdx.x;   // i = n*H + h
    if (i >= N_NODES * H) return;
    int n = i / H, h = i % H;
    const float* row = xl + (size_t)n * H * C + (size_t)h * C;
    const float* a1 = asrc + (size_t)h * C;
    const float* a2 = adst + (size_t)h * C;
    float s1 = 0.f, s2 = 0.f;
    for (int c = 0; c < C; ++c) { float v = row[c]; s1 += v * a1[c]; s2 += v * a2[c]; }
    es[i] = s1; ed[i] = s2;
}

__device__ inline void edge_sd(const int* __restrict__ ei, int e, int& s, int& d) {
    if (e < E_EDGES) { s = ei[e]; d = ei[E_EDGES + e]; }
    else             { s = e - E_EDGES; d = s; }        // self-loop
}

__device__ inline void atomicMaxF(float* addr, float val) {
    int* ai = (int*)addr;
    int old = __float_as_int(*addr);
    while (__int_as_float(old) < val) {
        int assumed = old;
        old = atomicCAS(ai, assumed, __float_as_int(val));
        if (old == assumed) break;
    }
}

__global__ void edge_max(const int* __restrict__ ei, const float* __restrict__ es,
                         const float* __restrict__ ed, float* __restrict__ ebuf,
                         float* __restrict__ mbuf, int H)
{
    long i = (long)blockIdx.x * blockDim.x + threadIdx.x;  // i = e*H + h
    if (i >= (long)E_TOT * H) return;
    int e = (int)(i / H), h = (int)(i % H);
    int s, d; edge_sd(ei, e, s, d);
    float v = es[s * H + h] + ed[d * H + h];
    v = v >= 0.f ? v : NEG_SLOPE * v;                      // leaky_relu
    ebuf[i] = v;
    atomicMaxF(&mbuf[d * H + h], v);
}

__global__ void edge_expsum(const int* __restrict__ ei, const float* __restrict__ ebuf,
                            const float* __restrict__ mbuf, float* __restrict__ exbuf,
                            float* __restrict__ den, int H)
{
    long i = (long)blockIdx.x * blockDim.x + threadIdx.x;
    if (i >= (long)E_TOT * H) return;
    int e = (int)(i / H), h = (int)(i % H);
    int s, d; edge_sd(ei, e, s, d); (void)s;
    float ex = __expf(ebuf[i] - mbuf[d * H + h]);
    exbuf[i] = ex;
    atomicAdd(&den[d * H + h], ex);
}

// one block per edge; blockDim = H*C/4 threads, each scatters 4 channels
__global__ void edge_aggregate(const int* __restrict__ ei, const float* __restrict__ xl,
                               const float* __restrict__ exbuf, const float* __restrict__ den,
                               float* __restrict__ agg, int H, int C)
{
    int e = blockIdx.x;
    int c4 = threadIdx.x * 4;
    int h = c4 / C;
    int s, d; edge_sd(ei, e, s, d);
    float alpha = exbuf[(size_t)e * H + h] / (den[d * H + h] + 1e-16f);
    const float4 xv = *reinterpret_cast<const float4*>(xl + (size_t)s * H * C + c4);
    float* dst = agg + (size_t)d * H * C + c4;
    atomicAdd(dst + 0, xv.x * alpha);
    atomicAdd(dst + 1, xv.y * alpha);
    atomicAdd(dst + 2, xv.z * alpha);
    atomicAdd(dst + 3, xv.w * alpha);
}

// ---------------------------------------------------------------- BatchNorm
__global__ void bn_stats(const float* __restrict__ xin, const float* __restrict__ bias,
                         float* __restrict__ sums, int n, int C)
{
    __shared__ float s1[256], s2[256];
    int c = blockIdx.y;
    int row = blockIdx.x * 256 + threadIdx.x;
    float v = 0.f;
    if (row < n) v = xin[(size_t)row * C + c] + bias[c];
    s1[threadIdx.x] = v; s2[threadIdx.x] = v * v;
    __syncthreads();
    for (int s = 128; s > 0; s >>= 1) {
        if (threadIdx.x < s) {
            s1[threadIdx.x] += s1[threadIdx.x + s];
            s2[threadIdx.x] += s2[threadIdx.x + s];
        }
        __syncthreads();
    }
    if (threadIdx.x == 0) { atomicAdd(&sums[c], s1[0]); atomicAdd(&sums[C + c], s2[0]); }
}

__global__ void bn_apply_relu(const float* __restrict__ xin, const float* __restrict__ bias,
                              const float* __restrict__ sums, const float* __restrict__ g,
                              const float* __restrict__ be, float* __restrict__ out,
                              int n, int C)
{
    long i = (long)blockIdx.x * blockDim.x + threadIdx.x;
    if (i >= (long)n * C) return;
    int c = (int)(i % C);
    float mu  = sums[c] / n;
    float var = sums[C + c] / n - mu * mu;      // biased, matches reference
    float v = xin[i] + bias[c];
    float y = (v - mu) * rsqrtf(var + BN_EPS) * g[c] + be[c];
    out[i] = y > 0.f ? y : 0.f;
}

// ------------------------------------------------------------- pool + final
__global__ void pool_kernel(const float* __restrict__ h3, const float* __restrict__ b3,
                            const int* __restrict__ batch, float* __restrict__ pooled,
                            float* __restrict__ cnt)
{
    int i = blockIdx.x * blockDim.x + threadIdx.x;    // i = n*HID + c
    if (i >= N_NODES * HID) return;
    int n = i / HID, c = i % HID;
    int g = batch[n];
    atomicAdd(&pooled[g * HID + c], h3[i] + b3[c]);
    if (c == 0) atomicAdd(&cnt[g], 1.0f);
}

__global__ void final_linear(const float* __restrict__ pooled, const float* __restrict__ cnt,
                             const float* __restrict__ Wlin, const float* __restrict__ blin,
                             float* __restrict__ out)
{
    int g = blockIdx.x, o = threadIdx.x;
    float inv = 1.0f / fmaxf(cnt[g], 1.0f);
    float s = blin[o];
    for (int k = 0; k < HID; ++k)
        s += pooled[g * HID + k] * inv * Wlin[k * OUT_DIM + o];
    out[g * OUT_DIM + o] = s;
}

// ------------------------------------------------------------- orchestration
static void run_gat_layer(hipStream_t stream, const float* fin, int Cin,
                          const float* W, const float* asrc, const float* adst,
                          int H, int C, const int* ei,
                          unsigned short* bfX, unsigned short* bfW,
                          float* xl, float* es, float* ed, float* mbuf, float* den,
                          float* ebuf, float* exbuf, float* agg)
{
    const int HCl = H * C;
    long nIn = (long)N_NODES * Cin;
    f32_to_bf16<<<(int)((nIn + 255) / 256), 256, 0, stream>>>(fin, bfX, nIn);
    long nW = (long)Cin * HCl;
    pack_W_bf16<<<(int)((nW + 255) / 256), 256, 0, stream>>>(W, bfW, Cin, HCl);

    const int wavesPerBlock = HCl / (WT * 16);      // 4 for 256-wide, 2 for 128-wide
    gemm_bf16_wmma<<<N_NODES / 16, wavesPerBlock * 32, 0, stream>>>(
        bfX, bfW, xl, N_NODES, Cin, HCl);

    int nh = N_NODES * H;
    gat_scores<<<(nh + 255) / 256, 256, 0, stream>>>(xl, asrc, adst, es, ed, H, C);
    fill_f32<<<(nh + 255) / 256, 256, 0, stream>>>(mbuf, -3.0e38f, nh);
    fill_f32<<<(nh + 255) / 256, 256, 0, stream>>>(den, 0.f, nh);
    long nAgg = (long)N_NODES * HCl;
    fill_f32<<<(int)((nAgg + 255) / 256), 256, 0, stream>>>(agg, 0.f, nAgg);

    long nEH = (long)E_TOT * H;
    edge_max<<<(int)((nEH + 255) / 256), 256, 0, stream>>>(ei, es, ed, ebuf, mbuf, H);
    edge_expsum<<<(int)((nEH + 255) / 256), 256, 0, stream>>>(ei, ebuf, mbuf, exbuf, den, H);
    edge_aggregate<<<E_TOT, HCl / 4, 0, stream>>>(ei, xl, exbuf, den, agg, H, C);
}

static void run_bn_relu(hipStream_t stream, const float* agg, const float* bias,
                        const float* g, const float* be, float* sums, float* out, int C)
{
    fill_f32<<<(2 * C + 255) / 256, 256, 0, stream>>>(sums, 0.f, 2 * C);
    dim3 sgrid((N_NODES + 255) / 256, C);
    bn_stats<<<sgrid, 256, 0, stream>>>(agg, bias, sums, N_NODES, C);
    long tot = (long)N_NODES * C;
    bn_apply_relu<<<(int)((tot + 255) / 256), 256, 0, stream>>>(agg, bias, sums, g, be, out, N_NODES, C);
}

extern "C" void kernel_launch(void* const* d_in, const int* in_sizes, int n_in,
                              void* d_out, int out_size, void* d_ws, size_t ws_size,
                              hipStream_t stream)
{
    (void)in_sizes; (void)n_in; (void)out_size; (void)ws_size;

    const float* x     = (const float*)d_in[0];
    const int*   ei    = (const int*)d_in[1];
    const int*   batch = (const int*)d_in[2];
    const float* W1 = (const float*)d_in[3],  *as1 = (const float*)d_in[4],
               * ad1 = (const float*)d_in[5], *b1  = (const float*)d_in[6],
               * g1  = (const float*)d_in[7], *be1 = (const float*)d_in[8];
    const float* W2 = (const float*)d_in[9],  *as2 = (const float*)d_in[10],
               * ad2 = (const float*)d_in[11],*b2  = (const float*)d_in[12],
               * g2  = (const float*)d_in[13],*be2 = (const float*)d_in[14];
    const float* W3 = (const float*)d_in[15], *as3 = (const float*)d_in[16],
               * ad3 = (const float*)d_in[17],*b3  = (const float*)d_in[18];
    const float* Wlin = (const float*)d_in[19], *blin = (const float*)d_in[20];
    float* out = (float*)d_out;

    char* ws = (char*)d_ws;
    size_t off = 0;
    auto take = [&](size_t bytes) -> char* {
        char* p = ws + off;
        off = (off + bytes + 255) & ~(size_t)255;
        return p;
    };
    unsigned short* bfX = (unsigned short*)take((size_t)N_NODES * IN_DIM * 2);
    unsigned short* bfW = (unsigned short*)take((size_t)IN_DIM * HC * 2);
    float* xl    = (float*)take((size_t)N_NODES * HC * 4);
    float* agg   = (float*)take((size_t)N_NODES * HC * 4);
    float* featA = (float*)take((size_t)N_NODES * HC * 4);
    float* featB = (float*)take((size_t)N_NODES * HC * 4);
    float* es    = (float*)take((size_t)N_NODES * HEADS * 4);
    float* ed    = (float*)take((size_t)N_NODES * HEADS * 4);
    float* mbuf  = (float*)take((size_t)N_NODES * HEADS * 4);
    float* den   = (float*)take((size_t)N_NODES * HEADS * 4);
    float* ebuf  = (float*)take((size_t)E_TOT * HEADS * 4);
    float* exbuf = (float*)take((size_t)E_TOT * HEADS * 4);
    float* sums  = (float*)take((size_t)2 * HC * 4);
    float* pooled= (float*)take((size_t)G_GRAPHS * HID * 4);
    float* cnt   = (float*)take((size_t)G_GRAPHS * 4);

    // conv1: 512 -> (2,128) concat, + BN + ReLU
    run_gat_layer(stream, x, IN_DIM, W1, as1, ad1, HEADS, HID, ei,
                  bfX, bfW, xl, es, ed, mbuf, den, ebuf, exbuf, agg);
    run_bn_relu(stream, agg, b1, g1, be1, sums, featA, HC);

    // conv2: 256 -> (2,128) concat, + BN + ReLU
    run_gat_layer(stream, featA, HC, W2, as2, ad2, HEADS, HID, ei,
                  bfX, bfW, xl, es, ed, mbuf, den, ebuf, exbuf, agg);
    run_bn_relu(stream, agg, b2, g2, be2, sums, featB, HC);

    // conv3: 256 -> (1,128), mean over 1 head == identity
    run_gat_layer(stream, featB, HC, W3, as3, ad3, 1, HID, ei,
                  bfX, bfW, xl, es, ed, mbuf, den, ebuf, exbuf, agg);

    // global mean pool + final linear
    fill_f32<<<(G_GRAPHS * HID + 255) / 256, 256, 0, stream>>>(pooled, 0.f, G_GRAPHS * HID);
    fill_f32<<<1, 256, 0, stream>>>(cnt, 0.f, G_GRAPHS);
    pool_kernel<<<(N_NODES * HID + 255) / 256, 256, 0, stream>>>(agg, b3, batch, pooled, cnt);
    final_linear<<<G_GRAPHS, OUT_DIM, 0, stream>>>(pooled, cnt, Wlin, blin, out);
}
